// MultitaskRNN_70643622084779
// MI455X (gfx1250) — matrix operand order repaired and checked by reading, not compiled, not run on hardware
//
#include <hip/hip_runtime.h>

// ---------------------------------------------------------------------------
// MultitaskRNN for MI455X (gfx1250, wave32, WMMA bf16 path)
//   B=64, T=512, I=128, H=512, O=64, ALPHA=0.1
//
// Pipeline:
//   prep_kernel   : Wcat = bf16([W_rec | W_in])  (H x 640), Wout->bf16, h0->bf16/f32
//   step_kernel(t): pre = [h_{t-1} | x_t] @ Wcatᵀ + b   (K = 512+128 = 640, WMMA)
//                   h_t = 0.9 h_{t-1} + 0.1 tanh(pre);  store f32 ping-pong + bf16 hist
//                   (512 launches; stream ordering = global sync between steps)
//   out_kernel    : out = hist @ W_outᵀ + b_out  ([B*T,512]x[512,64], WMMA)
//   hfinal_kernel : copy final f32 h into d_out tail
// ---------------------------------------------------------------------------

typedef __attribute__((ext_vector_type(16))) __bf16 v16bf;
typedef __attribute__((ext_vector_type(8)))  __bf16 v8bf;
typedef __attribute__((ext_vector_type(8)))  float  v8f;

#define B_  64
#define T_  512
#define I_  128
#define H_  512
#define O_  64
#define KC_ 640          // H_ + I_ concatenated K for the fused step GEMM

static __device__ __forceinline__ v8f wmma_bf16(v16bf a, v16bf b, v8f c) {
    // D = A(16x32 bf16) * B(32x16 bf16) + C(16x16 f32)
    return __builtin_amdgcn_wmma_f32_16x16x32_bf16(
        /*neg_a=*/false, a, /*neg_b=*/false, b,
        /*c_mod=*/(short)0, c, /*reuse_a=*/false, /*reuse_b=*/false);
}

// A fragment (16x32 bf16, row-major source, leading dim ld elements).
// lane<16: row=lane, K = {kb..kb+7, kb+16..kb+23}; lane>=16: +8 on both runs.
static __device__ __forceinline__ v16bf load_a_bf16(const __bf16* Abase, int ld,
                                                    int kb, int lane) {
    int row = lane & 15;
    int k0  = kb + ((lane >> 4) << 3);
    const v8bf* p0 = reinterpret_cast<const v8bf*>(Abase + row * ld + k0);
    const v8bf* p1 = reinterpret_cast<const v8bf*>(Abase + row * ld + k0 + 16);
    v8bf lo = *p0, hi = *p1;
    v16bf a;
#pragma unroll
    for (int j = 0; j < 8; ++j) { a[j] = lo[j]; a[8 + j] = hi[j]; }
    return a;
}

// Same A fragment but converting from an f32 row-major source on the fly.
static __device__ __forceinline__ v16bf load_a_f32(const float* Abase, int ld,
                                                   int kb, int lane) {
    int row = lane & 15;
    int k0  = kb + ((lane >> 4) << 3);
    const v8f* p0 = reinterpret_cast<const v8f*>(Abase + row * ld + k0);
    const v8f* p1 = reinterpret_cast<const v8f*>(Abase + row * ld + k0 + 16);
    v8f lo = *p0, hi = *p1;
    v16bf a;
#pragma unroll
    for (int j = 0; j < 8; ++j) {
        a[j]     = (__bf16)lo[j];
        a[8 + j] = (__bf16)hi[j];
    }
    return a;
}

// B fragment (32x16 bf16). Source is W row-major [N, K]; same K order as A:
// lane<16: K = {kb..kb+7, kb+16..kb+23}; lane>=16: +8 on both runs (two 16B loads).
static __device__ __forceinline__ v16bf load_b_bf16(const __bf16* Wtile, int ld,
                                                    int kb, int lane) {
    int col = lane & 15;
    int kk  = kb + ((lane >> 4) << 3);
    v8bf lo = *reinterpret_cast<const v8bf*>(Wtile + col * ld + kk);
    v8bf hi = *reinterpret_cast<const v8bf*>(Wtile + col * ld + kk + 16);
    v16bf b;
#pragma unroll
    for (int j = 0; j < 8; ++j) { b[j] = lo[j]; b[8 + j] = hi[j]; }
    return b;
}

// ---------------------------------------------------------------------------
__global__ __launch_bounds__(256) void prep_kernel(
    const float* __restrict__ W_rec, const float* __restrict__ W_in,
    const float* __restrict__ W_out, const float* __restrict__ hidden,
    unsigned short* __restrict__ Wcat_u, unsigned short* __restrict__ Wout_u,
    unsigned short* __restrict__ h0_u,  float* __restrict__ hbuf0) {
    __bf16* Wcat  = reinterpret_cast<__bf16*>(Wcat_u);
    __bf16* Woutb = reinterpret_cast<__bf16*>(Wout_u);
    __bf16* h0b   = reinterpret_cast<__bf16*>(h0_u);
    int idx = blockIdx.x * blockDim.x + threadIdx.x;
    int stride = gridDim.x * blockDim.x;
    for (int i = idx; i < H_ * KC_; i += stride) {
        int h = i / KC_, k = i - h * KC_;
        float v = (k < H_) ? W_rec[h * H_ + k] : W_in[h * I_ + (k - H_)];
        Wcat[i] = (__bf16)v;
    }
    for (int i = idx; i < O_ * H_; i += stride) Woutb[i] = (__bf16)W_out[i];
    for (int i = idx; i < B_ * H_; i += stride) {
        float v = hidden[i];
        h0b[i]   = (__bf16)v;
        hbuf0[i] = v;
    }
}

// ---------------------------------------------------------------------------
// One timestep. Grid: 16 blocks x 256 threads = 128 waves; each wave owns one
// 16x16 tile of pre[B_=64, H_=512]. K runs over 640 (512 from h_{t-1}, 128
// from x_t converted f32->bf16 on the fly).
__global__ __launch_bounds__(256) void step_kernel(
    const unsigned short* __restrict__ hprev_u, int hprev_ld,  // bf16, [B_, ld]
    const float* __restrict__ inputs, int t,
    const unsigned short* __restrict__ Wcat_u,                 // bf16 [H_, KC_]
    const float* __restrict__ bias,
    const float* __restrict__ hcur_f32,                        // [B_, H_]
    float* __restrict__ hnext_f32,                             // [B_, H_]
    unsigned short* __restrict__ hist_u)                       // bf16 [B_, T_, H_]
{
    const __bf16* hprev = reinterpret_cast<const __bf16*>(hprev_u);
    const __bf16* Wcat  = reinterpret_cast<const __bf16*>(Wcat_u);
    __bf16*       hist  = reinterpret_cast<__bf16*>(hist_u);

    const int lane = threadIdx.x & 31;
    const int wave = blockIdx.x * (blockDim.x >> 5) + (threadIdx.x >> 5);
    const int mt = wave >> 5;   // 0..3  : 16-row tile over B_
    const int nt = wave & 31;   // 0..31 : 16-col tile over H_

    const __bf16* Abase = hprev + (mt * 16) * hprev_ld;
    const __bf16* Wtile = Wcat + (nt * 16) * KC_;

    v8f c = {};
    // Recurrent part: K = 0..511 from h_{t-1} (bf16)
#pragma unroll
    for (int ks = 0; ks < H_ / 32; ++ks) {
        v16bf a  = load_a_bf16(Abase, hprev_ld, ks * 32, lane);
        v16bf bm = load_b_bf16(Wtile, KC_, ks * 32, lane);
        c = wmma_bf16(a, bm, c);
    }
    // Input part: K = 512..639 from inputs[:, t, :] (f32 -> bf16)
    const float* Ain = inputs + ((size_t)(mt * 16) * T_ + t) * I_;
#pragma unroll
    for (int ks = 0; ks < I_ / 32; ++ks) {
        v16bf a  = load_a_f32(Ain, T_ * I_, ks * 32, lane);
        v16bf bm = load_b_bf16(Wtile, KC_, H_ + ks * 32, lane);
        c = wmma_bf16(a, bm, c);
    }

    // Epilogue: leaky tanh update; f32 ping-pong + bf16 history row (t).
    const int col = lane & 15;
    const int n_g = nt * 16 + col;
    const float bn = bias[n_g];
#pragma unroll
    for (int v = 0; v < 8; ++v) {
        int m_g = mt * 16 + v + ((lane >> 4) << 3);
        float pre = c[v] + bn;
        float hp  = hcur_f32[m_g * H_ + n_g];
        float hn  = 0.9f * hp + 0.1f * tanhf(pre);
        hnext_f32[m_g * H_ + n_g] = hn;
        hist[((size_t)m_g * T_ + t) * H_ + n_g] = (__bf16)hn;
    }
}

// ---------------------------------------------------------------------------
// out[b,t,o] = hist[b,t,:] @ W_outᵀ + b_out : M=32768, N=64, K=512.
// 1024 blocks x 8 waves; one 16x16 tile per wave.
__global__ __launch_bounds__(256) void out_kernel(
    const unsigned short* __restrict__ hist_u,   // bf16 [B_*T_, H_]
    const unsigned short* __restrict__ Wout_u,   // bf16 [O_, H_]
    const float* __restrict__ b_out,
    float* __restrict__ out)                     // [B_*T_, O_]
{
    const __bf16* hist  = reinterpret_cast<const __bf16*>(hist_u);
    const __bf16* Woutb = reinterpret_cast<const __bf16*>(Wout_u);

    const int lane = threadIdx.x & 31;
    const int wave = blockIdx.x * (blockDim.x >> 5) + (threadIdx.x >> 5);
    const int mt = wave >> 2;  // 0..2047 over B_*T_
    const int nt = wave & 3;   // 0..3    over O_

    const __bf16* Abase = hist + (size_t)(mt * 16) * H_;
    const __bf16* Wtile = Woutb + (nt * 16) * H_;

    v8f c = {};
#pragma unroll
    for (int ks = 0; ks < H_ / 32; ++ks) {
        v16bf a  = load_a_bf16(Abase, H_, ks * 32, lane);
        v16bf bm = load_b_bf16(Wtile, H_, ks * 32, lane);
        c = wmma_bf16(a, bm, c);
    }

    const int col = lane & 15;
    const int n_g = nt * 16 + col;
    const float bo = b_out[n_g];
#pragma unroll
    for (int v = 0; v < 8; ++v) {
        int r = mt * 16 + v + ((lane >> 4) << 3);
        out[(size_t)r * O_ + n_g] = c[v] + bo;
    }
}

__global__ void hfinal_kernel(const float* __restrict__ hbuf,
                              float* __restrict__ dst) {
    int i = blockIdx.x * blockDim.x + threadIdx.x;
    if (i < B_ * H_) dst[i] = hbuf[i];
}

// ---------------------------------------------------------------------------
extern "C" void kernel_launch(void* const* d_in, const int* in_sizes, int n_in,
                              void* d_out, int out_size, void* d_ws, size_t ws_size,
                              hipStream_t stream) {
    const float* inputs = (const float*)d_in[0];   // [B,T,I]
    const float* hidden = (const float*)d_in[1];   // [B,H]
    const float* W_in   = (const float*)d_in[2];   // [H,I]
    const float* W_rec  = (const float*)d_in[3];   // [H,H]
    const float* bias   = (const float*)d_in[4];   // [H]
    const float* W_out  = (const float*)d_in[5];   // [O,H]
    const float* b_out  = (const float*)d_in[6];   // [O]
    float* out = (float*)d_out;

    char* ws = (char*)d_ws;
    auto alloc = [&](size_t bytes) -> char* {
        char* p = ws;
        ws += (bytes + 255) & ~(size_t)255;
        return p;
    };
    unsigned short* Wcat  = (unsigned short*)alloc((size_t)H_ * KC_ * 2);   // 640 KB
    unsigned short* Woutb = (unsigned short*)alloc((size_t)O_ * H_ * 2);    //  64 KB
    unsigned short* h0b   = (unsigned short*)alloc((size_t)B_ * H_ * 2);    //  64 KB
    float*          hbuf0 = (float*)alloc((size_t)B_ * H_ * 4);             // 128 KB
    float*          hbuf1 = (float*)alloc((size_t)B_ * H_ * 4);             // 128 KB
    unsigned short* hist  = (unsigned short*)alloc((size_t)B_ * T_ * H_ * 2); // 32 MB

    prep_kernel<<<256, 256, 0, stream>>>(W_rec, W_in, W_out, hidden,
                                         Wcat, Woutb, h0b, hbuf0);

    float* hb[2] = {hbuf0, hbuf1};
    for (int t = 0; t < T_; ++t) {
        // h_{t-1} as bf16: row m lives at hist[(m*T + t-1)*H] (ld = T*H);
        // for t==0 use the dedicated h0 buffer (ld = H).
        const unsigned short* hp = t ? hist + (size_t)(t - 1) * H_ : h0b;
        int ld = t ? T_ * H_ : H_;
        step_kernel<<<16, 256, 0, stream>>>(hp, ld, inputs, t, Wcat, bias,
                                            hb[t & 1], hb[(t + 1) & 1], hist);
    }

    out_kernel<<<1024, 256, 0, stream>>>(hist, Woutb, b_out, out);
    // T_ is even -> final f32 state sits in hbuf0.
    hfinal_kernel<<<(B_ * H_ + 255) / 256, 256, 0, stream>>>(
        hb[T_ & 1], out + (size_t)B_ * T_ * O_);
}
